// NetworkAwareHybridGNN_48893907697751
// MI455X (gfx1250) — matrix operand order repaired and checked by reading, not compiled, not run on hardware
//
#include <hip/hip_runtime.h>
#include <math.h>

// ---------------------------------------------------------------------------
// CDNA5 (gfx1250) implementation of NetworkAwareHybridGNN inference.
// GEMMs use v_wmma_f32_16x16x32_f16 (f16 in, f32 accumulate), 32x32 tile/wave
// (4 WMMAs per K-step, A and B fragments each reused twice). GEMM is
// specialized on (N,K) so row strides fold into 24-bit VMEM immediates.
// Edge softmax-aggregation uses f32 atomics (HBM/L2 bound phase).
// ---------------------------------------------------------------------------

typedef __attribute__((ext_vector_type(16))) _Float16 v16h;
typedef __attribute__((ext_vector_type(8)))  float    v8f;

#define THREADS 256

__device__ __forceinline__ void atomicMaxF(float* addr, float v) {
  if (v >= 0.0f) atomicMax((int*)addr, __float_as_int(v));
  else           atomicMin((unsigned int*)addr, __float_as_uint(v));
}

// A fragment 16x32 f16: lane L holds row (L&15); half i holds
// k = (i&7) + 8*(L>=16) + 16*(i>=8)   -> 4x float4 + convert
__device__ __forceinline__ v16h load_a_frag(const float* __restrict__ arow,
                                            int k0, int akb) {
  const float4* ap0 = (const float4*)(arow + k0 + akb);
  const float4* ap1 = (const float4*)(arow + k0 + akb + 16);
  float4 a0 = ap0[0], a1 = ap0[1], a2 = ap1[0], a3 = ap1[1];
  float av[16] = {a0.x, a0.y, a0.z, a0.w, a1.x, a1.y, a1.z, a1.w,
                  a2.x, a2.y, a2.z, a2.w, a3.x, a3.y, a3.z, a3.w};
  v16h f;
#pragma unroll
  for (int i = 0; i < 16; ++i) f[i] = (_Float16)av[i];
  return f;
}

// B fragment 32x16 f16: lane L holds col (L&15); half i holds
// k = (i&7) + 8*(L>=16) + 16*(i>=8)  (coalesced across the 16 lanes per row).
// With compile-time N the i*N stride folds into the VMEM immediate offset.
__device__ __forceinline__ v16h load_b_frag(const float* __restrict__ Wbase,
                                            long N) {
  v16h f;
#pragma unroll
  for (int i = 0; i < 16; ++i) f[i] = (_Float16)Wbase[(long)(i + (i >= 8 ? 8 : 0)) * N];
  return f;
}

// ---------------------------------------------------------------------------
// WMMA GEMM: out[M,N] = epilogue(A[M,K] @ W[K,N]); 32x32 tile per wave.
// NC/KC: compile-time N/K (0 = use runtime values).
// mode 0: raw   mode 1: +bias[n]   mode 2: relu(g[n]*(v+bias[n])*s + be[n])
// Requires M%32==0, N%32==0, K%32==0 (true for every GEMM here).
// ---------------------------------------------------------------------------
template <int NC, int KC>
__global__ __launch_bounds__(THREADS) void gemm_wmma_t(
    const float* __restrict__ A, const float* __restrict__ W,
    float* __restrict__ out, int M, int Nrt, int Krt,
    int mode, const float* __restrict__ bias,
    const float* __restrict__ gamma, const float* __restrict__ beta,
    float bnscale)
{
  const int N = (NC > 0) ? NC : Nrt;
  const int K = (KC > 0) ? KC : Krt;
  const int lane = threadIdx.x & 31;
  const int wave = threadIdx.x >> 5;
  const int Mt2 = M >> 5, Nt2 = N >> 5;
  const int t = blockIdx.x * (THREADS / 32) + wave;
  if (t >= Mt2 * Nt2) return;                    // wave-uniform guard
  const int tn2 = t % Nt2, tm2 = t / Nt2;        // waves in block share tm2 -> A reuse
  const int r16 = lane & 15;
  const int hi  = lane >> 4;                     // 0/1 lane half
  const int akb = hi * 8;
  const int bkb = hi * 8;
  const int m0 = tm2 * 32;
  const int n0 = tn2 * 32;
  const float* arow0 = A + (long)(m0 + r16) * K;       // rows m0..m0+15
  const float* arow1 = A + (long)(m0 + 16 + r16) * K;  // rows m0+16..m0+31
  const int ncol0 = n0 + r16;
  const int ncol1 = n0 + 16 + r16;

  // Precompute branchless epilogue: v -> v*scA + scB (+ optional relu)
  float scA0 = 1.0f, scB0 = 0.0f, scA1 = 1.0f, scB1 = 0.0f;
  int dorelu = 0;
  if (mode == 1) {
    scB0 = bias[ncol0];
    scB1 = bias[ncol1];
  } else if (mode == 2) {
    float g0 = gamma[ncol0] * bnscale, g1 = gamma[ncol1] * bnscale;
    scA0 = g0; scB0 = g0 * bias[ncol0] + beta[ncol0];
    scA1 = g1; scB1 = g1 * bias[ncol1] + beta[ncol1];
    dorelu = 1;
  }

  v8f acc00 = {}, acc01 = {}, acc10 = {}, acc11 = {};
  const float* wb0 = W + (long)bkb * N + ncol0;  // B col ncol0, k-rows bkb..bkb+7, bkb+16..bkb+23
  const float* wb1 = W + (long)bkb * N + ncol1;
  for (int k0 = 0; k0 < K; k0 += 32) {
    v16h a0 = load_a_frag(arow0, k0, akb);
    v16h a1 = load_a_frag(arow1, k0, akb);
    v16h b0 = load_b_frag(wb0, N);
    v16h b1 = load_b_frag(wb1, N);
    if (k0 + 32 < K) {  // prefetch next K-step of the strided B stream
      __builtin_prefetch(wb0 + (long)32 * N, 0, 1);
      __builtin_prefetch(wb0 + (long)40 * N, 0, 1);
    }
    wb0 += (long)32 * N;
    wb1 += (long)32 * N;
    acc00 = __builtin_amdgcn_wmma_f32_16x16x32_f16(false, a0, false, b0, (short)0, acc00, false, false);
    acc01 = __builtin_amdgcn_wmma_f32_16x16x32_f16(false, a0, false, b1, (short)0, acc01, false, false);
    acc10 = __builtin_amdgcn_wmma_f32_16x16x32_f16(false, a1, false, b0, (short)0, acc10, false, false);
    acc11 = __builtin_amdgcn_wmma_f32_16x16x32_f16(false, a1, false, b1, (short)0, acc11, false, false);
  }

#pragma unroll
  for (int j = 0; j < 8; ++j) {
    int ma = m0 + j + hi * 8;       // rows of acc0x
    int mb = ma + 16;               // rows of acc1x
    float v00 = acc00[j] * scA0 + scB0;
    float v01 = acc01[j] * scA1 + scB1;
    float v10 = acc10[j] * scA0 + scB0;
    float v11 = acc11[j] * scA1 + scB1;
    if (dorelu) {
      v00 = fmaxf(v00, 0.0f); v01 = fmaxf(v01, 0.0f);
      v10 = fmaxf(v10, 0.0f); v11 = fmaxf(v11, 0.0f);
    }
    out[(long)ma * N + ncol0] = v00;
    out[(long)ma * N + ncol1] = v01;
    out[(long)mb * N + ncol0] = v10;
    out[(long)mb * N + ncol1] = v11;
  }
}

// ---------------------------------------------------------------------------
// Helper kernels (GAT edge softmax-aggregation + small dense pieces)
// ---------------------------------------------------------------------------
__global__ __launch_bounds__(THREADS) void fill_kernel(float* __restrict__ p, long n, float v) {
  long i = (long)blockIdx.x * blockDim.x + threadIdx.x;
  if (i < n) p[i] = v;
}

// al_s[n,h] = sum_c xp[n,h,c]*a_src[h,c];  al_d similarly
__global__ __launch_bounds__(THREADS) void attn_node_kernel(
    const float* __restrict__ xp, const float* __restrict__ as_,
    const float* __restrict__ ad_, float* __restrict__ als,
    float* __restrict__ ald, int n, int Hh, int Cc)
{
  int i = blockIdx.x * blockDim.x + threadIdx.x;
  if (i >= n * Hh) return;
  int node = i / Hh, h = i % Hh;
  const float* row = xp + (long)node * (Hh * Cc) + h * Cc;
  float s = 0.0f, d = 0.0f;
  for (int c = 0; c < Cc; ++c) {
    float x = row[c];
    s += x * as_[h * Cc + c];
    d += x * ad_[h * Cc + c];
  }
  als[i] = s; ald[i] = d;
}

// ve[k,h] = sum_c We[k, h*C+c] * ae[h,c]   (folds edge projection into 6xH)
__global__ void ve_kernel(const float* __restrict__ We, const float* __restrict__ ae,
                          float* __restrict__ ve, int Hh, int Cc)
{
  int i = threadIdx.x;
  if (i >= 6 * Hh) return;
  int k = i / Hh, h = i % Hh;
  float s = 0.0f;
  for (int c = 0; c < Cc; ++c) s += We[k * (Hh * Cc) + h * Cc + c] * ae[h * Cc + c];
  ve[k * Hh + h] = s;
}

// alpha = leaky_relu(al_s[src]+al_d[dst]+al_e, 0.2); segment max into amax[dst]
__global__ __launch_bounds__(THREADS) void edge_alpha_kernel(
    const int* __restrict__ src, const int* __restrict__ dst,
    const float* __restrict__ ea, const float* __restrict__ ve,
    const float* __restrict__ als, const float* __restrict__ ald,
    float* __restrict__ alpha, float* __restrict__ amax, int Ee, int Hh)
{
  int i = blockIdx.x * blockDim.x + threadIdx.x;
  if (i >= Ee * Hh) return;
  int e = i / Hh, h = i % Hh;
  int s = src[e], d = dst[e];
  float a = als[s * Hh + h] + ald[d * Hh + h];
#pragma unroll
  for (int k = 0; k < 6; ++k) a += ea[e * 6 + k] * ve[k * Hh + h];
  a = a > 0.0f ? a : 0.2f * a;
  alpha[i] = a;
  atomicMaxF(&amax[d * Hh + h], a);
}

// ex = exp(alpha - amax[dst]); segment sum into den[dst]; alpha <- ex
__global__ __launch_bounds__(THREADS) void edge_ex_kernel(
    const int* __restrict__ dst, float* __restrict__ alpha,
    const float* __restrict__ amax, float* __restrict__ den, int Ee, int Hh)
{
  int i = blockIdx.x * blockDim.x + threadIdx.x;
  if (i >= Ee * Hh) return;
  int e = i / Hh, h = i % Hh;
  float ex = __expf(alpha[i] - amax[dst[e] * Hh + h]);
  alpha[i] = ex;
  atomicAdd(&den[dst[e] * Hh + h], ex);
}

// alpha <- ex / (den[dst] + 1e-16)
__global__ __launch_bounds__(THREADS) void edge_w_kernel(
    const int* __restrict__ dst, float* __restrict__ alpha,
    const float* __restrict__ den, int Ee, int Hh)
{
  int i = blockIdx.x * blockDim.x + threadIdx.x;
  if (i >= Ee * Hh) return;
  int e = i / Hh, h = i % Hh;
  alpha[i] = alpha[i] / (den[dst[e] * Hh + h] + 1e-16f);
}

// out[dst, :] += xp[src, :] * w[e,h]  (float4 gather, 4 scatter atomics/thread)
__global__ __launch_bounds__(THREADS) void edge_scatter_kernel(
    const int* __restrict__ src, const int* __restrict__ dst,
    const float* __restrict__ xp, const float* __restrict__ w,
    float* __restrict__ out, int Ee, int Hh, int Cc)
{
  int per = (Hh * Cc) >> 2;  // channel groups of 4 (== 64 for D=256)
  long i = (long)blockIdx.x * blockDim.x + threadIdx.x;
  if (i >= (long)Ee * per) return;
  int e = (int)(i / per), cg = (int)(i % per);
  int c0 = cg * 4;
  int h = c0 / Cc;
  float wt = w[e * Hh + h];
  int s = src[e], d = dst[e];
  const float4 v = *(const float4*)(xp + (long)s * (Hh * Cc) + c0);
  float* o = out + (long)d * (Hh * Cc) + c0;
  atomicAdd(o + 0, v.x * wt);
  atomicAdd(o + 1, v.y * wt);
  atomicAdd(o + 2, v.z * wt);
  atomicAdd(o + 3, v.w * wt);
}

// out = (out + bias), optional relu (in place)
__global__ __launch_bounds__(THREADS) void finish_kernel(
    float* __restrict__ out, const float* __restrict__ bias, long n, int Dd, int relu)
{
  long i = (long)blockIdx.x * blockDim.x + threadIdx.x;
  if (i >= n) return;
  int dcol = (int)(i % Dd);
  float v = out[i] + bias[dcol];
  if (relu) v = v > 0.0f ? v : 0.0f;
  out[i] = v;
}

// h1 = relu(bn(nf @ w1 + b1))  with K=10 (too small for WMMA)
__global__ __launch_bounds__(THREADS) void net1_kernel(
    const float* __restrict__ nf, const float* __restrict__ w1,
    const float* __restrict__ b1, const float* __restrict__ g1,
    const float* __restrict__ be1, float* __restrict__ h1,
    int Bb, int Dd, float s)
{
  int i = blockIdx.x * blockDim.x + threadIdx.x;
  if (i >= Bb * Dd) return;
  int b = i / Dd, n = i % Dd;
  float acc = 0.0f;
#pragma unroll
  for (int k = 0; k < 10; ++k) acc += nf[b * 10 + k] * w1[k * Dd + n];
  float v = g1[n] * (acc + b1[n]) * s + be1[n];
  h1[i] = v > 0.0f ? v : 0.0f;
}

// combined[b] = concat(x3[ids[b]], net_emb[b])
__global__ __launch_bounds__(THREADS) void concat_kernel(
    const float* __restrict__ x3, const float* __restrict__ nemb,
    const int* __restrict__ ids, float* __restrict__ comb, int Bb, int Dd)
{
  int i = blockIdx.x * blockDim.x + threadIdx.x;
  if (i >= Bb * 2 * Dd) return;
  int b = i / (2 * Dd), d = i % (2 * Dd);
  comb[i] = (d < Dd) ? x3[(long)ids[b] * Dd + d] : nemb[b * Dd + (d - Dd)];
}

// ---------------------------------------------------------------------------
// Host orchestration
// ---------------------------------------------------------------------------
static inline int cdiv_l(long a, long b) { return (int)((a + b - 1) / b); }

extern "C" void kernel_launch(void* const* d_in, const int* in_sizes, int n_in,
                              void* d_out, int out_size, void* d_ws, size_t ws_size,
                              hipStream_t stream) {
  const int* ids       = (const int*)d_in[0];
  const float* nf      = (const float*)d_in[1];
  const int* ei        = (const int*)d_in[2];
  const float* ea      = (const float*)d_in[3];
  const float* emb     = (const float*)d_in[4];
  const float* net_w1  = (const float*)d_in[5];
  const float* net_b1  = (const float*)d_in[6];
  const float* net_g1  = (const float*)d_in[7];
  const float* net_be1 = (const float*)d_in[8];
  const float* net_w2  = (const float*)d_in[9];
  const float* net_b2  = (const float*)d_in[10];
  const float* net_g2  = (const float*)d_in[11];
  const float* net_be2 = (const float*)d_in[12];
  const float* fus_w   = (const float*)d_in[31];
  const float* fus_b   = (const float*)d_in[32];
  const float* fus_g   = (const float*)d_in[33];
  const float* fus_be  = (const float*)d_in[34];
  const float* cls_w   = (const float*)d_in[35];
  const float* cls_b   = (const float*)d_in[36];

  const int D  = 256;
  const int Bn = in_sizes[0];          // 2048
  const int Ee = in_sizes[2] / 2;      // 320000
  const int Nn = in_sizes[4] / D;      // 20000
  const int NCLS = in_sizes[36];       // 20000
  const int* src = ei;
  const int* dst = ei + Ee;

  // workspace carve-up (16-float aligned so float4 loads stay 16B aligned)
  float* w = (float*)d_ws;
  auto carve = [&](size_t nfl) { float* p = w; w += ((nfl + 15) & ~(size_t)15); return p; };
  float* bufA  = carve((size_t)Nn * D);   // xp scratch
  float* bufB  = carve((size_t)Nn * D);   // x1 / x3
  float* bufC  = carve((size_t)Nn * D);   // x2
  float* als   = carve((size_t)Nn * 8);
  float* ald   = carve((size_t)Nn * 8);
  float* amax  = carve((size_t)Nn * 8);
  float* den   = carve((size_t)Nn * 8);
  float* alpha = carve((size_t)Ee * 8);   // alpha -> ex -> w (in place)
  float* ve    = carve(64);
  float* h1    = carve((size_t)Bn * D);
  float* nemb  = carve((size_t)Bn * D);
  float* comb  = carve((size_t)Bn * 2 * D);
  float* fused = carve((size_t)Bn * D);

  const float BNS = 0.9999950000374997f;  // 1/sqrt(1 + 1e-5)

  auto gemm = [&](const float* A, const float* Wm, float* O, int M, int N, int K,
                  int mode, const float* bias, const float* g, const float* be) {
    long tiles = (long)(M / 32) * (N / 32);
    int blocks = cdiv_l(tiles, THREADS / 32);
    if (N == 256 && K == 256)
      gemm_wmma_t<256, 256><<<blocks, THREADS, 0, stream>>>(A, Wm, O, M, N, K, mode, bias, g, be, BNS);
    else if (N == 256 && K == 512)
      gemm_wmma_t<256, 512><<<blocks, THREADS, 0, stream>>>(A, Wm, O, M, N, K, mode, bias, g, be, BNS);
    else if (N == 20000 && K == 256)
      gemm_wmma_t<20000, 256><<<blocks, THREADS, 0, stream>>>(A, Wm, O, M, N, K, mode, bias, g, be, BNS);
    else
      gemm_wmma_t<0, 0><<<blocks, THREADS, 0, stream>>>(A, Wm, O, M, N, K, mode, bias, g, be, BNS);
  };

  auto gat = [&](const float* xin, float* xpbuf, float* outbuf, int Hh,
                 const float* Wm, const float* as_, const float* ad_,
                 const float* Wem, const float* ae_, const float* bias, int relu) {
    const int Cc = D / Hh;
    gemm(xin, Wm, xpbuf, Nn, D, D, 0, nullptr, nullptr, nullptr);
    attn_node_kernel<<<cdiv_l((long)Nn * Hh, THREADS), THREADS, 0, stream>>>(
        xpbuf, as_, ad_, als, ald, Nn, Hh, Cc);
    ve_kernel<<<1, 64, 0, stream>>>(Wem, ae_, ve, Hh, Cc);
    fill_kernel<<<cdiv_l((long)Nn * Hh, THREADS), THREADS, 0, stream>>>(amax, (long)Nn * Hh, -INFINITY);
    fill_kernel<<<cdiv_l((long)Nn * Hh, THREADS), THREADS, 0, stream>>>(den, (long)Nn * Hh, 0.0f);
    fill_kernel<<<cdiv_l((long)Nn * D, THREADS), THREADS, 0, stream>>>(outbuf, (long)Nn * D, 0.0f);
    edge_alpha_kernel<<<cdiv_l((long)Ee * Hh, THREADS), THREADS, 0, stream>>>(
        src, dst, ea, ve, als, ald, alpha, amax, Ee, Hh);
    edge_ex_kernel<<<cdiv_l((long)Ee * Hh, THREADS), THREADS, 0, stream>>>(
        dst, alpha, amax, den, Ee, Hh);
    edge_w_kernel<<<cdiv_l((long)Ee * Hh, THREADS), THREADS, 0, stream>>>(
        dst, alpha, den, Ee, Hh);
    edge_scatter_kernel<<<cdiv_l((long)Ee * (D / 4), THREADS), THREADS, 0, stream>>>(
        src, dst, xpbuf, alpha, outbuf, Ee, Hh, Cc);
    finish_kernel<<<cdiv_l((long)Nn * D, THREADS), THREADS, 0, stream>>>(
        outbuf, bias, (long)Nn * D, D, relu);
  };

  // Three GAT layers (H=8, H=8, H=1)
  gat(emb,  bufA, bufB, 8, (const float*)d_in[13], (const float*)d_in[14],
      (const float*)d_in[15], (const float*)d_in[16], (const float*)d_in[17],
      (const float*)d_in[18], 1);
  gat(bufB, bufA, bufC, 8, (const float*)d_in[19], (const float*)d_in[20],
      (const float*)d_in[21], (const float*)d_in[22], (const float*)d_in[23],
      (const float*)d_in[24], 1);
  gat(bufC, bufA, bufB, 1, (const float*)d_in[25], (const float*)d_in[26],
      (const float*)d_in[27], (const float*)d_in[28], (const float*)d_in[29],
      (const float*)d_in[30], 0);

  // Network-feature MLP
  net1_kernel<<<cdiv_l((long)Bn * D, THREADS), THREADS, 0, stream>>>(
      nf, net_w1, net_b1, net_g1, net_be1, h1, Bn, D, BNS);
  gemm(h1, net_w2, nemb, Bn, D, D, 2, net_b2, net_g2, net_be2);

  // Fuse node embedding (gathered) with network embedding
  concat_kernel<<<cdiv_l((long)Bn * 2 * D, THREADS), THREADS, 0, stream>>>(
      bufB, nemb, ids, comb, Bn, D);
  gemm(comb, fus_w, fused, Bn, D, 2 * D, 2, fus_b, fus_g, fus_be);

  // Classifier: [2048,256] @ [256,20000] + bias  -> d_out
  gemm(fused, cls_w, (float*)d_out, Bn, NCLS, D, 1, cls_b, nullptr, nullptr);
}
